// GNN12_46093589020762
// MI455X (gfx1250) — compile-verified, matrix operand order checked by reading, not run on hardware
//
#include <hip/hip_runtime.h>
#include <hip/hip_bf16.h>

#define N_NODES 65536
#define N_EDGES 1048576
#define IN_FEATS 128
#define H1 128
#define H2 256
#define N_CLASSES 16

typedef __attribute__((ext_vector_type(16))) __bf16 bf16x16;
typedef __attribute__((ext_vector_type(8)))  __bf16 bf16x8;
typedef __attribute__((ext_vector_type(4)))  __bf16 bf16x4;
typedef __attribute__((ext_vector_type(8)))  float  f32x8;

// ---- fragment builders -----------------------------------------------------
// 16-bit WMMA operand layout: per lane, halves 0..7 = K khalf..khalf+7,
// halves 8..15 = K 16+khalf..16+khalf+7 (khalf = 0 for lanes 0-15, 8 for 16-31)

__device__ __forceinline__ bf16x16 frag_bf16(const __bf16* p0, const __bf16* p1) {
  bf16x8 lo = *(const bf16x8*)p0;   // 16B load
  bf16x8 hi = *(const bf16x8*)p1;   // 16B load
  return __builtin_shufflevector(lo, hi, 0,1,2,3,4,5,6,7,8,9,10,11,12,13,14,15);
}

__device__ __forceinline__ bf16x16 frag_f32(const float* p0, const float* p1) {
  float4 a0 = *(const float4*)(p0);
  float4 a1 = *(const float4*)(p0 + 4);
  float4 b0 = *(const float4*)(p1);
  float4 b1 = *(const float4*)(p1 + 4);
  bf16x16 f;
  f[0]  = (__bf16)a0.x; f[1]  = (__bf16)a0.y; f[2]  = (__bf16)a0.z; f[3]  = (__bf16)a0.w;
  f[4]  = (__bf16)a1.x; f[5]  = (__bf16)a1.y; f[6]  = (__bf16)a1.z; f[7]  = (__bf16)a1.w;
  f[8]  = (__bf16)b0.x; f[9]  = (__bf16)b0.y; f[10] = (__bf16)b0.z; f[11] = (__bf16)b0.w;
  f[12] = (__bf16)b1.x; f[13] = (__bf16)b1.y; f[14] = (__bf16)b1.z; f[15] = (__bf16)b1.w;
  return f;
}

__device__ __forceinline__ f32x8 wmma_bf16(bf16x16 a, bf16x16 b, f32x8 c) {
  return __builtin_amdgcn_wmma_f32_16x16x32_bf16(false, a, false, b, (short)0, c,
                                                 false, false);
}

__device__ __forceinline__ float leaky(float t) { return fmaxf(t, 0.01f * t); }

// ---- async global->LDS staging (CDNA5, ASYNCcnt-tracked) -------------------
__device__ __forceinline__ void async_copy16(unsigned lds_byte_off, const __bf16* gsrc) {
  asm volatile("global_load_async_to_lds_b128 %0, %1, off"
               :: "v"(lds_byte_off), "v"(gsrc) : "memory");
}
__device__ __forceinline__ void async_wait0() {
  asm volatile("s_wait_asynccnt 0x0" ::: "memory");
}
__device__ __forceinline__ void stage_lds(__bf16* lds, const __bf16* g, int nelem, int tid) {
  for (int c = tid; c < (nelem >> 3); c += 256)
    async_copy16((unsigned)(size_t)&lds[c * 8], g + c * 8);
  async_wait0();
  __syncthreads();
}

// ---------------------------------------------------------------------------
// Preprocess: f32 -> bf16 (n % 4 == 0), vectorized 4-wide
// ---------------------------------------------------------------------------
__global__ void __launch_bounds__(256) cvt_bf16_kernel(const float* __restrict__ in,
                                                       __bf16* __restrict__ out, int n) {
  int i = (blockIdx.x * 256 + threadIdx.x) * 4;
  if (i < n) {
    float4 v = *(const float4*)(in + i);
    bf16x4 o;
    o[0] = (__bf16)v.x; o[1] = (__bf16)v.y; o[2] = (__bf16)v.z; o[3] = (__bf16)v.w;
    *(bf16x4*)(out + i) = o;
  }
}

// ---------------------------------------------------------------------------
// Kernel A: m = relu(x[src] @ Wp.T + bp), pooled[dst] = max(pooled[dst], m)
// Wave = 2 tiles x 16 edges x 128 out-features; block = 8 waves = 256 edges.
// B fragment fetched once per (nt,kb) and reused by both tiles (1 ds per wmma).
// ---------------------------------------------------------------------------
__global__ void __launch_bounds__(256) edge_mlp_scatter_max(
    const __bf16* __restrict__ xb, const __bf16* __restrict__ Wpb,
    const float* __restrict__ bp, const int* __restrict__ src,
    const int* __restrict__ dst, float* __restrict__ pooled) {
  __shared__ __bf16 sW[H1 * IN_FEATS];  // 32 KB
  int tid = threadIdx.x;
  stage_lds(sW, Wpb, H1 * IN_FEATS, tid);

  int wave  = tid >> 5;
  int lane  = tid & 31;
  int mrow  = lane & 15;
  int khalf = (lane & 16) ? 8 : 0;
  int hi    = (lane & 16) ? 8 : 0;
  int ebase = blockIdx.x * 256 + wave * 32;   // two 16-edge tiles per wave

  const __bf16* xr0 = xb + (unsigned)src[ebase + mrow] * IN_FEATS;
  const __bf16* xr1 = xb + (unsigned)src[ebase + 16 + mrow] * IN_FEATS;
  bf16x16 A0[4], A1[4];
  #pragma unroll
  for (int kb = 0; kb < 4; ++kb) {
    A0[kb] = frag_bf16(xr0 + kb * 32 + khalf, xr0 + kb * 32 + 16 + khalf);
    A1[kb] = frag_bf16(xr1 + kb * 32 + khalf, xr1 + kb * 32 + 16 + khalf);
  }

  int didx0[8], didx1[8];
  #pragma unroll
  for (int v = 0; v < 8; ++v) {
    didx0[v] = dst[ebase + hi + v];
    didx1[v] = dst[ebase + 16 + hi + v];
  }

  #pragma unroll
  for (int nt = 0; nt < 8; ++nt) {
    const __bf16* wr = sW + (nt * 16 + mrow) * IN_FEATS;
    f32x8 acc0 = {}, acc1 = {};
    #pragma unroll
    for (int kb = 0; kb < 4; ++kb) {
      bf16x16 B = frag_bf16(wr + kb * 32 + khalf, wr + kb * 32 + 16 + khalf);
      acc0 = wmma_bf16(A0[kb], B, acc0);
      acc1 = wmma_bf16(A1[kb], B, acc1);
    }
    float bias = bp[nt * 16 + mrow];
    unsigned col = (unsigned)(nt * 16 + mrow);
    #pragma unroll
    for (int v = 0; v < 8; ++v) {
      // relu => non-negative, so u32 max == f32 max on the bit patterns
      float r0 = fmaxf(acc0[v] + bias, 0.f);
      float r1 = fmaxf(acc1[v] + bias, 0.f);
      atomicMax((unsigned int*)pooled + (unsigned)didx0[v] * (unsigned)H1 + col,
                __float_as_uint(r0));
      atomicMax((unsigned int*)pooled + (unsigned)didx1[v] * (unsigned)H1 + col,
                __float_as_uint(r1));
    }
  }
}

// ---------------------------------------------------------------------------
// Kernel B: h1 = leaky_relu(x @ Ws.T + pooled @ Wn.T + bn)  -> bf16
// ---------------------------------------------------------------------------
__global__ void __launch_bounds__(256) combine_kernel(
    const __bf16* __restrict__ xb, const float* __restrict__ pooled,
    const __bf16* __restrict__ Wsb, const __bf16* __restrict__ Wnb,
    const float* __restrict__ bn, __bf16* __restrict__ h1b) {
  __shared__ __bf16 sWs[H1 * IN_FEATS];  // 32 KB
  __shared__ __bf16 sWn[H1 * H1];        // 32 KB
  int tid = threadIdx.x;
  for (int c = tid; c < (H1 * IN_FEATS) >> 3; c += 256) {
    async_copy16((unsigned)(size_t)&sWs[c * 8], Wsb + c * 8);
    async_copy16((unsigned)(size_t)&sWn[c * 8], Wnb + c * 8);
  }
  async_wait0();
  __syncthreads();

  int wave = tid >> 5, lane = tid & 31;
  int mrow = lane & 15, khalf = (lane & 16) ? 8 : 0, hi = (lane & 16) ? 8 : 0;
  unsigned rbase = blockIdx.x * 128 + wave * 16;

  const __bf16* xr = xb + (rbase + mrow) * IN_FEATS;
  const float*  pr = pooled + (rbase + mrow) * H1;
  bf16x16 AX[4], AP[4];
  #pragma unroll
  for (int kb = 0; kb < 4; ++kb) {
    AX[kb] = frag_bf16(xr + kb * 32 + khalf, xr + kb * 32 + 16 + khalf);
    AP[kb] = frag_f32(pr + kb * 32 + khalf, pr + kb * 32 + 16 + khalf);
  }

  #pragma unroll
  for (int nt = 0; nt < 8; ++nt) {
    const __bf16* wsr = sWs + (nt * 16 + mrow) * IN_FEATS;
    const __bf16* wnr = sWn + (nt * 16 + mrow) * H1;
    f32x8 acc = {};
    #pragma unroll
    for (int kb = 0; kb < 4; ++kb) {
      acc = wmma_bf16(AX[kb], frag_bf16(wsr + kb * 32 + khalf, wsr + kb * 32 + 16 + khalf), acc);
      acc = wmma_bf16(AP[kb], frag_bf16(wnr + kb * 32 + khalf, wnr + kb * 32 + 16 + khalf), acc);
    }
    float bias = bn[nt * 16 + mrow];
    unsigned col = (unsigned)(nt * 16 + mrow);
    #pragma unroll
    for (int v = 0; v < 8; ++v)
      h1b[(rbase + hi + v) * H1 + col] = (__bf16)leaky(acc[v] + bias);
  }
}

// ---------------------------------------------------------------------------
// Kernel C: h2 = leaky_relu(h1 @ W1.T + b1)  (W1: [256,128]) -> bf16
// ---------------------------------------------------------------------------
__global__ void __launch_bounds__(256) mlp1_kernel(
    const __bf16* __restrict__ h1b, const __bf16* __restrict__ W1b,
    const float* __restrict__ b1, __bf16* __restrict__ h2b) {
  __shared__ __bf16 sW[H2 * H1];  // 64 KB
  int tid = threadIdx.x;
  stage_lds(sW, W1b, H2 * H1, tid);

  int wave = tid >> 5, lane = tid & 31;
  int mrow = lane & 15, khalf = (lane & 16) ? 8 : 0, hi = (lane & 16) ? 8 : 0;
  unsigned rbase = blockIdx.x * 128 + wave * 16;

  const __bf16* hr = h1b + (rbase + mrow) * H1;
  bf16x16 A[4];
  #pragma unroll
  for (int kb = 0; kb < 4; ++kb)
    A[kb] = frag_bf16(hr + kb * 32 + khalf, hr + kb * 32 + 16 + khalf);

  #pragma unroll
  for (int nt = 0; nt < 16; ++nt) {
    const __bf16* wr = sW + (nt * 16 + mrow) * H1;
    f32x8 acc = {};
    #pragma unroll
    for (int kb = 0; kb < 4; ++kb)
      acc = wmma_bf16(A[kb], frag_bf16(wr + kb * 32 + khalf, wr + kb * 32 + 16 + khalf), acc);
    float bias = b1[nt * 16 + mrow];
    unsigned col = (unsigned)(nt * 16 + mrow);
    #pragma unroll
    for (int v = 0; v < 8; ++v)
      h2b[(rbase + hi + v) * H2 + col] = (__bf16)leaky(acc[v] + bias);
  }
}

// ---------------------------------------------------------------------------
// Kernel D: out = sigmoid(h2 @ W2.T + b2)  (W2: [16,256]) -> f32
// ---------------------------------------------------------------------------
__global__ void __launch_bounds__(256) mlp2_kernel(
    const __bf16* __restrict__ h2b, const __bf16* __restrict__ W2b,
    const float* __restrict__ b2, float* __restrict__ out) {
  __shared__ __bf16 sW[N_CLASSES * H2];  // 8 KB
  int tid = threadIdx.x;
  stage_lds(sW, W2b, N_CLASSES * H2, tid);

  int wave = tid >> 5, lane = tid & 31;
  int mrow = lane & 15, khalf = (lane & 16) ? 8 : 0, hi = (lane & 16) ? 8 : 0;
  unsigned rbase = blockIdx.x * 128 + wave * 16;

  const __bf16* hr = h2b + (rbase + mrow) * H2;
  const __bf16* wr = sW + mrow * H2;  // B column n = mrow -> row n of W2
  f32x8 acc = {};
  #pragma unroll
  for (int kb = 0; kb < 8; ++kb) {
    bf16x16 a = frag_bf16(hr + kb * 32 + khalf, hr + kb * 32 + 16 + khalf);
    bf16x16 b = frag_bf16(wr + kb * 32 + khalf, wr + kb * 32 + 16 + khalf);
    acc = wmma_bf16(a, b, acc);
  }
  float bias = b2[mrow];
  #pragma unroll
  for (int v = 0; v < 8; ++v) {
    float t = acc[v] + bias;
    t = 1.0f / (1.0f + __expf(-t));
    out[(rbase + hi + v) * N_CLASSES + mrow] = t;
  }
}

extern "C" void kernel_launch(void* const* d_in, const int* in_sizes, int n_in,
                              void* d_out, int out_size, void* d_ws, size_t ws_size,
                              hipStream_t stream) {
  (void)in_sizes; (void)n_in; (void)out_size; (void)ws_size;
  const float* x  = (const float*)d_in[0];
  const float* Wp = (const float*)d_in[1];
  const float* bp = (const float*)d_in[2];
  const float* Ws = (const float*)d_in[3];
  const float* Wn = (const float*)d_in[4];
  const float* bn = (const float*)d_in[5];
  const float* W1 = (const float*)d_in[6];
  const float* b1 = (const float*)d_in[7];
  const float* W2 = (const float*)d_in[8];
  const float* b2 = (const float*)d_in[9];
  const int* src  = (const int*)d_in[10];
  const int* dst  = (const int*)d_in[11];

  char* ws = (char*)d_ws;
  size_t off = 0;
  float*  pooled = (float*)(ws + off);  off += (size_t)N_NODES * H1 * 4;        // 32 MB
  __bf16* xb     = (__bf16*)(ws + off); off += (size_t)N_NODES * IN_FEATS * 2;  // 16 MB
  __bf16* h1b    = (__bf16*)(ws + off); off += (size_t)N_NODES * H1 * 2;        // 16 MB
  __bf16* h2b    = (__bf16*)(ws + off); off += (size_t)N_NODES * H2 * 2;        // 32 MB
  __bf16* Wpb    = (__bf16*)(ws + off); off += (size_t)H1 * IN_FEATS * 2;
  __bf16* Wsb    = (__bf16*)(ws + off); off += (size_t)H1 * IN_FEATS * 2;
  __bf16* Wnb    = (__bf16*)(ws + off); off += (size_t)H1 * H1 * 2;
  __bf16* W1b    = (__bf16*)(ws + off); off += (size_t)H2 * H1 * 2;
  __bf16* W2b    = (__bf16*)(ws + off); off += (size_t)N_CLASSES * H2 * 2;

  hipMemsetAsync(pooled, 0, (size_t)N_NODES * H1 * sizeof(float), stream);

  // f32 -> bf16 preprocessing (one-shot; removes per-use cvt chains)
  cvt_bf16_kernel<<<(N_NODES * IN_FEATS / 4 + 255) / 256, 256, 0, stream>>>(x, xb, N_NODES * IN_FEATS);
  cvt_bf16_kernel<<<(H1 * IN_FEATS / 4 + 255) / 256, 256, 0, stream>>>(Wp, Wpb, H1 * IN_FEATS);
  cvt_bf16_kernel<<<(H1 * IN_FEATS / 4 + 255) / 256, 256, 0, stream>>>(Ws, Wsb, H1 * IN_FEATS);
  cvt_bf16_kernel<<<(H1 * H1 / 4 + 255) / 256, 256, 0, stream>>>(Wn, Wnb, H1 * H1);
  cvt_bf16_kernel<<<(H2 * H1 / 4 + 255) / 256, 256, 0, stream>>>(W1, W1b, H2 * H1);
  cvt_bf16_kernel<<<(N_CLASSES * H2 / 4 + 255) / 256, 256, 0, stream>>>(W2, W2b, N_CLASSES * H2);

  edge_mlp_scatter_max<<<N_EDGES / 256, 256, 0, stream>>>(xb, Wpb, bp, src, dst, pooled);
  combine_kernel<<<N_NODES / 128, 256, 0, stream>>>(xb, pooled, Wsb, Wnb, bn, h1b);
  mlp1_kernel<<<N_NODES / 128, 256, 0, stream>>>(h1b, W1b, b1, h2b);
  mlp2_kernel<<<N_NODES / 128, 256, 0, stream>>>(h2b, W2b, b2, (float*)d_out);
}
